// KPFCNN_35648228556895
// MI455X (gfx1250) — compile-verified, matrix operand order checked by reading, not run on hardware
//
#include <hip/hip_runtime.h>

// ---------------- CDNA5 types ----------------
typedef __attribute__((ext_vector_type(16))) __bf16 v16bf;
typedef __attribute__((ext_vector_type(8)))  float  v8f;

#define KP      15
#define LEAKY   0.1f
#define GN_EPS  1e-5f

// ---------------- device helpers ----------------
__device__ __forceinline__ __bf16 f2bf(float f) {
    unsigned u = __float_as_uint(f);
    u += 0x7fffu + ((u >> 16) & 1u);          // round-to-nearest-even
    unsigned short h = (unsigned short)(u >> 16);
    __bf16 r; __builtin_memcpy(&r, &h, 2);
    return r;
}
__device__ __forceinline__ __bf16 bf_zero() {
    unsigned short h = 0; __bf16 r; __builtin_memcpy(&r, &h, 2); return r;
}

// ---------------- fp32 -> bf16 cast ----------------
__global__ __launch_bounds__(256) void k_f32_to_bf16(const float* __restrict__ s,
                                                     __bf16* __restrict__ d,
                                                     long long n) {
    long long i = (long long)blockIdx.x * 256 + threadIdx.x;
    long long stride = (long long)gridDim.x * 256;
    for (; i < n; i += stride) d[i] = f2bf(s[i]);
}

// ---------------- bf16 WMMA GEMM: C[M,N] = A[M,K] * B[K,N] ----------------
// Block tile 128x128, 8 waves (wave32); wave tile 32x64 = 8 WMMAs per K-step.
// A staged row-major [row][k]; B staged TRANSPOSED [col][k] (padded) so both
// operand fragments are two contiguous 16B LDS segments -> ds_load_b128.
#define BM  128
#define BN  128
#define BK  32
#define BKP 40      // padded k-stride for sBt: 80B rows, 16B aligned, bank-friendly

__global__ __launch_bounds__(256) void k_gemm_bf16(const __bf16* __restrict__ A,
                                                   const __bf16* __restrict__ B,
                                                   float* __restrict__ C,
                                                   int M, int N, int K) {
    __shared__ __bf16 sA[BM * BK];     // [row][k]
    __shared__ __bf16 sBt[BN * BKP];   // [col][k] padded

    const int tid  = threadIdx.x;
    const int lane = tid & 31;
    const int wave = tid >> 5;
    const int wm   = wave >> 1;          // 0..3 : 32-row strip
    const int wn   = wave & 1;           // 0..1 : 64-col strip
    const int rowBase = blockIdx.y * BM;
    const int colBase = blockIdx.x * BN;

    const int lrow    = lane & 15;
    const int kbase   = (lane >> 4) * 8;   // 0 / 8
    const int halfsel = lane >> 4;

    // staging maps
    const int arow  = tid >> 1;            // 0..127
    const int acol0 = (tid & 1) * 16;      // 0 / 16
    const int bk0   = (tid >> 4) * 2;      // 0..30 step 2
    const int bc0   = (tid & 15) * 8;      // 0..120 step 8

    v8f acc[2][4] = {};

    union Frag { v16bf v; uint4 u[2]; };

    for (int kt = 0; kt < K; kt += BK) {
        // ---- stage A tile (128x32), 16 elems / thread ----
        {
            const int gr = rowBase + arow;
            const bool fullK = ((K & 7) == 0) && (kt + BK <= K);
            if (fullK && gr < M) {
                const uint4* p = (const uint4*)(A + (size_t)gr * K + kt + acol0);
                *(uint4*)&sA[arow * BK + acol0]     = p[0];
                *(uint4*)&sA[arow * BK + acol0 + 8] = p[1];
            } else {
#pragma unroll
                for (int j = 0; j < 16; ++j) {
                    int c = acol0 + j;
                    sA[arow * BK + c] =
                        (gr < M && kt + c < K) ? A[(size_t)gr * K + kt + c] : bf_zero();
                }
            }
        }
        // ---- stage B tile (32x128) transposed into sBt, 16 elems / thread ----
        {
            const bool fullN = ((N & 7) == 0) && (colBase + BN <= N);
#pragma unroll
            for (int rr = 0; rr < 2; ++rr) {
                int k  = bk0 + rr;
                int gk = kt + k;
                __bf16 tmp[8];
                if (fullN && gk < K) {
                    uint4 v = *(const uint4*)(B + (size_t)gk * N + colBase + bc0);
                    __builtin_memcpy(tmp, &v, 16);
                } else {
#pragma unroll
                    for (int j = 0; j < 8; ++j) {
                        int gc = colBase + bc0 + j;
                        tmp[j] = (gk < K && gc < N) ? B[(size_t)gk * N + gc] : bf_zero();
                    }
                }
#pragma unroll
                for (int j = 0; j < 8; ++j) sBt[(bc0 + j) * BKP + k] = tmp[j];
            }
        }
        __syncthreads();

        if (kt + BK < K)   // hint next A tile toward the caches
            __builtin_prefetch(&A[(size_t)(rowBase + (tid >> 1)) * K + kt + BK], 0, 1);

        // ---- fragments: two 16B LDS segments each (e0-7 = k kbase.., e8-15 = kbase+16..) ----
        v16bf afr[2], bfr[4];
#pragma unroll
        for (int mi = 0; mi < 2; ++mi) {
            const __bf16* pa = &sA[(wm * 32 + mi * 16 + lrow) * BK];
            Frag f;
            f.u[0] = *(const uint4*)(pa + kbase);
            f.u[1] = *(const uint4*)(pa + kbase + 16);
            afr[mi] = f.v;
        }
#pragma unroll
        for (int ni = 0; ni < 4; ++ni) {
            const __bf16* pb = &sBt[(wn * 64 + ni * 16 + lrow) * BKP];
            Frag f;
            f.u[0] = *(const uint4*)(pb + kbase);
            f.u[1] = *(const uint4*)(pb + kbase + 16);
            bfr[ni] = f.v;
        }
#pragma unroll
        for (int mi = 0; mi < 2; ++mi)
#pragma unroll
            for (int ni = 0; ni < 4; ++ni)
                acc[mi][ni] = __builtin_amdgcn_wmma_f32_16x16x32_bf16(
                    false, afr[mi], false, bfr[ni], (short)0, acc[mi][ni], false, false);
        __syncthreads();
    }

    // ---- store: VGPR r -> rows r (lanes 0-15) / r+8 (lanes 16-31), col = lane&15 ----
#pragma unroll
    for (int mi = 0; mi < 2; ++mi)
#pragma unroll
        for (int ni = 0; ni < 4; ++ni)
#pragma unroll
            for (int r = 0; r < 8; ++r) {
                int grow = rowBase + wm * 32 + mi * 16 + r + halfsel * 8;
                int gcol = colBase + wn * 64 + ni * 16 + lrow;
                if (grow < M && gcol < N)
                    C[(size_t)grow * N + gcol] = acc[mi][ni][r];
            }
}

// ---------------- fused [bias] + GroupNorm(8) + [LeakyReLU], in place ----------------
__global__ __launch_bounds__(256) void k_bias_gn_act(float* __restrict__ x,
                                                     const float* __restrict__ bias,
                                                     const float* __restrict__ gamma,
                                                     const float* __restrict__ beta,
                                                     int C, float slope,
                                                     int act, int hasBias) {
    __shared__ float gsum[8], gsq[8];
    const int tid = threadIdx.x;
    const size_t base = (size_t)blockIdx.x * C;
    const int gs = C >> 3;                // group size (C divisible by 8)
    if (tid < 8) { gsum[tid] = 0.f; gsq[tid] = 0.f; }
    __syncthreads();
    for (int c = tid; c < C; c += 256) {
        float v = x[base + c] + (hasBias ? bias[c] : 0.f);
        x[base + c] = v;
        int g = c / gs;
        atomicAdd(&gsum[g], v);
        atomicAdd(&gsq[g], v * v);
    }
    __syncthreads();
    for (int c = tid; c < C; c += 256) {
        int g = c / gs;
        float mean = gsum[g] / gs;
        float var  = gsq[g] / gs - mean * mean;
        float rs   = rsqrtf(var + GN_EPS);
        float v = (x[base + c] - mean) * rs * gamma[c] + beta[c];
        if (act) v = (v > 0.f) ? v : slope * v;
        x[base + c] = v;
    }
}

// ---------------- KPConv aggregation: agg[m, p*C+c] = sum_k w(m,k,p)*f[idx,c] ----------------
__global__ __launch_bounds__(256) void k_kpconv_agg(const float* __restrict__ qp,
                                                    const float* __restrict__ sp,
                                                    const float* __restrict__ feats,
                                                    const int*   __restrict__ nbr,
                                                    const float* __restrict__ kpts,
                                                    float sigma, int Ns, int Kn, int C,
                                                    __bf16* __restrict__ agg) {
    __shared__ float sw[40 * KP];
    __shared__ int   sidx[40];
    __shared__ float sq[3];
    const int m = blockIdx.x;
    const int tid = threadIdx.x;
    if (tid < 3) sq[tid] = qp[(size_t)m * 3 + tid];
    if (tid < Kn) sidx[tid] = nbr[(size_t)m * Kn + tid];
    __syncthreads();
    for (int t = tid; t < Kn * KP; t += 256) {
        int k = t / KP, p = t % KP;
        int id = sidx[k];
        float w = 0.f;
        if (id >= 0 && id < Ns) {
            float rx = sp[(size_t)id * 3 + 0] - sq[0] - kpts[p * 3 + 0];
            float ry = sp[(size_t)id * 3 + 1] - sq[1] - kpts[p * 3 + 1];
            float rz = sp[(size_t)id * 3 + 2] - sq[2] - kpts[p * 3 + 2];
            float d2 = rx * rx + ry * ry + rz * rz;
            w = fmaxf(1.f - sqrtf(d2 + 1e-12f) / sigma, 0.f);
        }
        sw[t] = w;
    }
    __syncthreads();
    for (int c = tid; c < C; c += 256) {
        float accv[KP];
#pragma unroll
        for (int p = 0; p < KP; ++p) accv[p] = 0.f;
        for (int k = 0; k < Kn; ++k) {
            int id = sidx[k];
            if (id >= 0 && id < Ns) {
                float f = feats[(size_t)id * C + c];
#pragma unroll
                for (int p = 0; p < KP; ++p) accv[p] += sw[k * KP + p] * f;
            }
        }
        size_t ob = (size_t)m * (KP * C) + c;
#pragma unroll
        for (int p = 0; p < KP; ++p) agg[ob + (size_t)p * C] = f2bf(accv[p]);
    }
}

// ---------------- maxpool over neighbors (shadow rows contribute 0) ----------------
__global__ __launch_bounds__(256) void k_maxpool(const float* __restrict__ f,
                                                 const int* __restrict__ idx,
                                                 float* __restrict__ out,
                                                 int Ns, int Kn, int C) {
    const int m = blockIdx.x;
    for (int c = threadIdx.x; c < C; c += 256) {
        float acc = -3.0e38f;
        for (int k = 0; k < Kn; ++k) {
            int id = idx[(size_t)m * Kn + k];
            float v = (id >= 0 && id < Ns) ? f[(size_t)id * C + c] : 0.f;
            acc = fmaxf(acc, v);
        }
        out[(size_t)m * C + c] = acc;
    }
}

// ---------------- nearest-neighbor upsample (idx[:,0]) ----------------
__global__ __launch_bounds__(256) void k_nearest(const float* __restrict__ f,
                                                 const int* __restrict__ idx,
                                                 float* __restrict__ out,
                                                 long long total, int Ns, int Kn, int C) {
    long long i = (long long)blockIdx.x * 256 + threadIdx.x;
    long long stride = (long long)gridDim.x * 256;
    for (; i < total; i += stride) {
        long long m = i / C; int c = (int)(i % C);
        int id = idx[(size_t)m * Kn];
        out[i] = (id >= 0 && id < Ns) ? f[(size_t)id * C + c] : 0.f;
    }
}

// ---------------- concat along channels ----------------
__global__ __launch_bounds__(256) void k_concat(const float* __restrict__ a,
                                                const float* __restrict__ b,
                                                float* __restrict__ out,
                                                long long total, int Ca, int Cb) {
    long long i = (long long)blockIdx.x * 256 + threadIdx.x;
    long long stride = (long long)gridDim.x * 256;
    int C = Ca + Cb;
    for (; i < total; i += stride) {
        long long m = i / C; int c = (int)(i % C);
        out[i] = (c < Ca) ? a[(size_t)m * Ca + c] : b[(size_t)m * Cb + (c - Ca)];
    }
}

// ---------------- residual add + leaky ----------------
__global__ __launch_bounds__(256) void k_add_leaky(const float* __restrict__ a,
                                                   const float* __restrict__ b,
                                                   float* __restrict__ out,
                                                   long long n, float slope) {
    long long i = (long long)blockIdx.x * 256 + threadIdx.x;
    long long stride = (long long)gridDim.x * 256;
    for (; i < n; i += stride) {
        float v = a[i] + b[i];
        out[i] = (v > 0.f) ? v : slope * v;
    }
}

// ---------------- bias add (classifier head) ----------------
__global__ __launch_bounds__(256) void k_bias(float* __restrict__ x,
                                              const float* __restrict__ b,
                                              long long n, int C) {
    long long i = (long long)blockIdx.x * 256 + threadIdx.x;
    long long stride = (long long)gridDim.x * 256;
    for (; i < n; i += stride) x[i] += b[i % C];
}

// =================================================================================
//                                   host side
// =================================================================================
namespace {
struct Arena {
    char* base; size_t off; size_t cap;
    void* alloc(size_t n) {
        size_t a = (off + 255) & ~(size_t)255;
        off = a + n;
        return (void*)(base + a);
    }
};
struct Lin  { const float *b, *beta, *gamma, *w; int ci, co; };
struct Conv { const float *W, *beta, *gamma, *kp; int ci, co; };
struct Res  { Conv cv; Lin sc; Lin u1; Lin u2; bool hasSc; int ci, co; };
} // namespace

extern "C" void kernel_launch(void* const* d_in, const int* in_sizes, int n_in,
                              void* d_out, int out_size, void* d_ws, size_t ws_size,
                              hipStream_t stream) {
    (void)n_in; (void)out_size;
    const float* feats = (const float*)d_in[0];

    // ---- walk the 202 param leaves in jax tree-flatten (sorted-key) order ----
    int pi = 1;
    auto nxt = [&]() { return (const float*)d_in[pi++]; };
    auto getLin  = [&](int ci, int co) { Lin l; l.b = nxt(); l.beta = nxt(); l.gamma = nxt(); l.w = nxt(); l.ci = ci; l.co = co; return l; };
    auto getConv = [&](int ci, int co) { Conv c; c.W = nxt(); c.beta = nxt(); c.gamma = nxt(); c.kp = nxt(); c.ci = ci; c.co = co; return c; };
    auto getRes  = [&](int ci, int co) {
        Res r; int h = co / 4;
        r.cv = getConv(h, h);                       // 'conv'
        r.hasSc = (ci != co);
        if (r.hasSc) r.sc = getLin(ci, co);         // 'shortcut'
        r.u1 = getLin(ci, h);                       // 'unary1'
        r.u2 = getLin(h, co);                       // 'unary2'
        r.ci = ci; r.co = co; return r;
    };

    Lin cls1 = getLin(128, 64);
    const float* cls2_b = nxt();
    const float* cls2_w = nxt();
    Lin dec1 = getLin(384, 128);
    Lin dec2 = getLin(768, 256);
    Lin dec3 = getLin(1536, 512);
    Lin dec4 = getLin(3072, 1024);
    Conv enc1_1 = getConv(5, 64);
    Res enc1_2 = getRes(64, 128);
    Res enc2_1 = getRes(128, 128);
    Res enc2_2 = getRes(128, 256);
    Res enc2_3 = getRes(256, 256);
    Res enc3_1 = getRes(256, 256);
    Res enc3_2 = getRes(256, 512);
    Res enc3_3 = getRes(512, 512);
    Res enc4_1 = getRes(512, 512);
    Res enc4_2 = getRes(512, 1024);
    Res enc4_3 = getRes(1024, 1024);
    Res enc5_1 = getRes(1024, 1024);
    Res enc5_2 = getRes(1024, 2048);
    Res enc5_3 = getRes(2048, 2048);

    // ---- pyramid tensors ----
    const float* pts[5]; int NP[5];
    for (int i = 0; i < 5; ++i) { pts[i] = (const float*)d_in[pi]; NP[i] = in_sizes[pi] / 3; ++pi; }
    const int* nbr[5]; int Knb[5];
    for (int i = 0; i < 5; ++i) { nbr[i] = (const int*)d_in[pi]; Knb[i] = in_sizes[pi] / NP[i]; ++pi; }
    const int* sub[4]; int Ksub[4];
    for (int i = 0; i < 4; ++i) { sub[i] = (const int*)d_in[pi]; Ksub[i] = in_sizes[pi] / NP[i + 1]; ++pi; }
    const int* up[4]; int Kup[4];
    for (int i = 0; i < 4; ++i) { up[i] = (const int*)d_in[pi]; Kup[i] = in_sizes[pi] / NP[i]; ++pi; }

    const float SIG[5] = {0.08f, 0.16f, 0.32f, 0.64f, 1.28f};

    Arena per{ (char*)d_ws, 0, ws_size };

    auto blocks1d = [](long long n) { long long b = (n + 255) / 256; return (int)(b > 1048576 ? 1048576 : b); };

    auto toBf = [&](const float* s, __bf16* d, size_t n) {
        k_f32_to_bf16<<<blocks1d((long long)n), 256, 0, stream>>>(s, d, (long long)n);
    };
    auto gemmRaw = [&](const __bf16* A, const __bf16* B, float* C, int M, int N, int K) {
        dim3 g((N + BN - 1) / BN, (M + BM - 1) / BM);
        k_gemm_bf16<<<g, 256, 0, stream>>>(A, B, C, M, N, K);
    };
    auto gemmF = [&](const float* A, const float* B, float* C, int M, int N, int K) {
        size_t mk = per.off;
        __bf16* Ab = (__bf16*)per.alloc((size_t)M * K * 2);
        __bf16* Bb = (__bf16*)per.alloc((size_t)K * N * 2);
        toBf(A, Ab, (size_t)M * K);
        toBf(B, Bb, (size_t)K * N);
        gemmRaw(Ab, Bb, C, M, N, K);
        per.off = mk;                 // stream-ordered reuse is safe (single stream)
    };
    auto gnAct = [&](float* x, const float* b, const float* gamma, const float* beta,
                     int N, int C, float slope, bool act, bool hasB) {
        k_bias_gn_act<<<N, 256, 0, stream>>>(x, b, gamma, beta, C, slope, act ? 1 : 0, hasB ? 1 : 0);
    };

    auto convBlock = [&](const float* qp, int M, const float* sp, int Ns,
                         const float* f, int Cin, const int* idx, int Kn,
                         const Conv& cv, float sigma) -> float* {
        float* out = (float*)per.alloc((size_t)M * cv.co * 4);
        size_t mk = per.off;
        __bf16* agg = (__bf16*)per.alloc((size_t)M * KP * Cin * 2);
        k_kpconv_agg<<<M, 256, 0, stream>>>(qp, sp, f, idx, cv.kp, sigma, Ns, Kn, Cin, agg);
        __bf16* Wb = (__bf16*)per.alloc((size_t)KP * Cin * cv.co * 2);
        toBf(cv.W, Wb, (size_t)KP * Cin * cv.co);
        gemmRaw(agg, Wb, out, M, cv.co, KP * Cin);
        gnAct(out, nullptr, cv.gamma, cv.beta, M, cv.co, LEAKY, true, false);
        per.off = mk;
        return out;
    };

    auto resBlock = [&](const float* qp, int M, const float* sp, int Ns,
                        const float* f, const int* idx, int Kn,
                        const Res& p, float sigma, bool strided) -> float* {
        float* out = (float*)per.alloc((size_t)M * p.co * 4);
        size_t mk = per.off;
        int h = p.cv.ci;
        // unary1 (on support features)
        float* x1 = (float*)per.alloc((size_t)Ns * h * 4);
        gemmF(f, p.u1.w, x1, Ns, h, p.ci);
        gnAct(x1, p.u1.b, p.u1.gamma, p.u1.beta, Ns, h, LEAKY, true, true);
        // KPConv
        float* x2 = convBlock(qp, M, sp, Ns, x1, h, idx, Kn, p.cv, sigma);
        // unary2 (no activation)
        float* x3 = (float*)per.alloc((size_t)M * p.co * 4);
        gemmF(x2, p.u2.w, x3, M, p.co, h);
        gnAct(x3, p.u2.b, p.u2.gamma, p.u2.beta, M, p.co, LEAKY, false, true);
        // shortcut
        const float* sc = f;
        if (strided) {
            float* mp = (float*)per.alloc((size_t)M * p.ci * 4);
            k_maxpool<<<M, 256, 0, stream>>>(f, idx, mp, Ns, Kn, p.ci);
            sc = mp;
        }
        if (p.hasSc) {
            float* s2 = (float*)per.alloc((size_t)M * p.co * 4);
            gemmF(sc, p.sc.w, s2, M, p.co, p.sc.ci);
            gnAct(s2, p.sc.b, p.sc.gamma, p.sc.beta, M, p.co, LEAKY, false, true);
            sc = s2;
        }
        long long n = (long long)M * p.co;
        k_add_leaky<<<blocks1d(n), 256, 0, stream>>>(x3, sc, out, n, LEAKY);
        per.off = mk;
        return out;
    };

    auto decode = [&](const float* fUp, int NsUp, int Cup, const int* upIdx, int Ku,
                      const float* fSkip, int M, int Cskip, const Lin& p) -> float* {
        float* out = (float*)per.alloc((size_t)M * p.co * 4);
        size_t mk = per.off;
        float* ni = (float*)per.alloc((size_t)M * Cup * 4);
        long long tn = (long long)M * Cup;
        k_nearest<<<blocks1d(tn), 256, 0, stream>>>(fUp, upIdx, ni, tn, NsUp, Ku, Cup);
        float* cat = (float*)per.alloc((size_t)M * (Cup + Cskip) * 4);
        long long tc = (long long)M * (Cup + Cskip);
        k_concat<<<blocks1d(tc), 256, 0, stream>>>(ni, fSkip, cat, tc, Cup, Cskip);
        gemmF(cat, p.w, out, M, p.co, p.ci);
        gnAct(out, p.b, p.gamma, p.beta, M, p.co, LEAKY, true, true);
        per.off = mk;
        return out;
    };

    // ---------------- encoder ----------------
    float* f1 = convBlock(pts[0], NP[0], pts[0], NP[0], feats, 5, nbr[0], Knb[0], enc1_1, SIG[0]);
    f1 = resBlock(pts[0], NP[0], pts[0], NP[0], f1, nbr[0], Knb[0], enc1_2, SIG[0], false);

    float* f2 = resBlock(pts[1], NP[1], pts[0], NP[0], f1, sub[0], Ksub[0], enc2_1, SIG[0], true);
    f2 = resBlock(pts[1], NP[1], pts[1], NP[1], f2, nbr[1], Knb[1], enc2_2, SIG[1], false);
    f2 = resBlock(pts[1], NP[1], pts[1], NP[1], f2, nbr[1], Knb[1], enc2_3, SIG[1], false);

    float* f3 = resBlock(pts[2], NP[2], pts[1], NP[1], f2, sub[1], Ksub[1], enc3_1, SIG[1], true);
    f3 = resBlock(pts[2], NP[2], pts[2], NP[2], f3, nbr[2], Knb[2], enc3_2, SIG[2], false);
    f3 = resBlock(pts[2], NP[2], pts[2], NP[2], f3, nbr[2], Knb[2], enc3_3, SIG[2], false);

    float* f4 = resBlock(pts[3], NP[3], pts[2], NP[2], f3, sub[2], Ksub[2], enc4_1, SIG[2], true);
    f4 = resBlock(pts[3], NP[3], pts[3], NP[3], f4, nbr[3], Knb[3], enc4_2, SIG[3], false);
    f4 = resBlock(pts[3], NP[3], pts[3], NP[3], f4, nbr[3], Knb[3], enc4_3, SIG[3], false);

    float* f5 = resBlock(pts[4], NP[4], pts[3], NP[3], f4, sub[3], Ksub[3], enc5_1, SIG[3], true);
    f5 = resBlock(pts[4], NP[4], pts[4], NP[4], f5, nbr[4], Knb[4], enc5_2, SIG[4], false);
    f5 = resBlock(pts[4], NP[4], pts[4], NP[4], f5, nbr[4], Knb[4], enc5_3, SIG[4], false);

    // ---------------- decoder ----------------
    float* l4 = decode(f5, NP[4], 2048, up[3], Kup[3], f4, NP[3], 1024, dec4);
    float* l3 = decode(l4, NP[3], 1024, up[2], Kup[2], f3, NP[2], 512, dec3);
    float* l2 = decode(l3, NP[2], 512,  up[1], Kup[1], f2, NP[1], 256, dec2);
    float* l1 = decode(l2, NP[1], 256,  up[0], Kup[0], f1, NP[0], 128, dec1);

    // ---------------- classifier head ----------------
    {
        float* h = (float*)per.alloc((size_t)NP[0] * 64 * 4);
        size_t mk = per.off;
        gemmF(l1, cls1.w, h, NP[0], 64, 128);
        gnAct(h, cls1.b, cls1.gamma, cls1.beta, NP[0], 64, 0.0f, true, true);   // slope 0 => ReLU
        gemmF(h, cls2_w, (float*)d_out, NP[0], 10, 64);
        long long n = (long long)NP[0] * 10;
        k_bias<<<blocks1d(n), 256, 0, stream>>>((float*)d_out, cls2_b, n, 10);
        per.off = mk;
    }
}
